// DynamoLoss_83648783057448
// MI455X (gfx1250) — matrix operand order repaired and checked by reading, not compiled, not run on hardware
//
#include <hip/hip_runtime.h>

typedef _Float16 v16h __attribute__((ext_vector_type(16)));
typedef _Float16 v8h  __attribute__((ext_vector_type(8)));
typedef _Float16 v4h  __attribute__((ext_vector_type(4)));
typedef float    v8f  __attribute__((ext_vector_type(8)));
typedef float    v4f  __attribute__((ext_vector_type(4)));
typedef float    v2f  __attribute__((ext_vector_type(2)));

#define B_N   8192
#define E_N   8
#define H_N   128
#define TILE  128
#define BSTR  136   // padded LDS row stride in halves (272B: 16B aligned, breaks bank alignment)

#define SIM_T  0.8f
#define EFF_T  0.1f
#define W_LB   0.1f
#define W_EFF  0.05f
#define W_CONS 0.1f
#define W_ENT  0.01f
#define W_TEMP 0.01f

// ---------------- workspace layout (bytes) ----------------
// 0   : float acc[12]   (0=task,1=eff,2=ent,3..10=usage[8])
// 48  : double kl_sum
// 56  : u64 pair_cnt  (includes diagonal; corrected by -B_N in finalize)
// 256 : float a[B]                (32768)
// 33024 : float logp[B][8]        (262144)
// 295168: float p[B][8]           (262144)
// 557312: _Float16 nrm16[B][128]  (2097152)  -> total ~2.6 MB

__global__ void zero_acc_kernel(unsigned int* acc32) {
    if (threadIdx.x < 32) acc32[threadIdx.x] = 0u;   // zeroes first 128 bytes
}

__global__ __launch_bounds__(256)
void prep_kernel(const float* __restrict__ logits, const int* __restrict__ targets,
                 const float* __restrict__ rp, const float* __restrict__ emb,
                 float* __restrict__ acc, float* __restrict__ a_out,
                 float* __restrict__ logp_out, float* __restrict__ p_out,
                 _Float16* __restrict__ nrm16)
{
    __shared__ float red[8][11];     // per-wave partials: 0=task,1=eff,2=ent,3..10=usage

    const int wavein = threadIdx.x >> 5;
    const int row    = (blockIdx.x * blockDim.x + threadIdx.x) >> 5;  // one wave32 per row
    const int lane   = threadIdx.x & 31;

    // ---- normalize embedding row (H=128 -> 4 floats/lane) ----
    v4f x = *(const v4f*)(emb + (size_t)row * H_N + lane * 4);
    float ss = x[0]*x[0] + x[1]*x[1] + x[2]*x[2] + x[3]*x[3];
    #pragma unroll
    for (int m = 16; m > 0; m >>= 1) ss += __shfl_xor(ss, m, 32);
    const float rinv = rsqrtf(ss);
    v4h hv;
    hv[0] = (_Float16)(x[0] * rinv); hv[1] = (_Float16)(x[1] * rinv);
    hv[2] = (_Float16)(x[2] * rinv); hv[3] = (_Float16)(x[3] * rinv);
    *(v4h*)(nrm16 + (size_t)row * H_N + lane * 4) = hv;

    if (lane == 0) {
        // ---- routing stats: logp = log_softmax(rp), p = exp(logp), a = sum p*logp ----
        float v[E_N], mx = -1e30f;
        #pragma unroll
        for (int e = 0; e < E_N; ++e) { v[e] = rp[(size_t)row * E_N + e]; mx = fmaxf(mx, v[e]); }
        float z = 0.f;
        #pragma unroll
        for (int e = 0; e < E_N; ++e) z += __expf(v[e] - mx);
        const float lse = mx + __logf(z);
        float a = 0.f, eff = 0.f, ent = 0.f;
        #pragma unroll
        for (int e = 0; e < E_N; ++e) {
            const float lp = v[e] - lse;
            const float pe = __expf(lp);
            logp_out[(size_t)row * E_N + e] = lp;
            p_out[(size_t)row * E_N + e]    = pe;
            a += pe * lp;
            if (v[e] < EFF_T) eff += v[e];
            ent -= v[e] * __logf(v[e] + 1e-8f);
            red[wavein][3 + e] = v[e];
        }
        a_out[row] = a;
        red[wavein][1] = eff;
        red[wavein][2] = ent;

        // ---- task cross-entropy (3 classes) ----
        const int t = targets[row];
        const float l0 = logits[(size_t)row * 3 + 0];
        const float l1 = logits[(size_t)row * 3 + 1];
        const float l2 = logits[(size_t)row * 3 + 2];
        const float mm = fmaxf(fmaxf(l0, l1), l2);
        const float lz = mm + __logf(__expf(l0 - mm) + __expf(l1 - mm) + __expf(l2 - mm));
        const float lt = (t == 0) ? l0 : ((t == 1) ? l1 : l2);
        red[wavein][0] = lz - lt;
    }
    __syncthreads();
    if (threadIdx.x < 11) {          // one atomic per accumulator per block (8x fewer RMWs)
        float s = 0.f;
        #pragma unroll
        for (int w = 0; w < 8; ++w) s += red[w][threadIdx.x];
        atomicAdd(&acc[threadIdx.x], s);
    }
}

__global__ __launch_bounds__(256)
void cons_kernel(const _Float16* __restrict__ nrm16, const float* __restrict__ logp,
                 const float* __restrict__ p, const float* __restrict__ a,
                 double* __restrict__ kl_acc, unsigned long long* __restrict__ cnt_acc)
{
    __shared__ _Float16 Bs[TILE][BSTR];      // j-tile of nrm (row-major, padded)
    __shared__ float logp_s[TILE][E_N];      // i-rows
    __shared__ float p_s[TILE][E_N];         // j-rows
    __shared__ float a_s[TILE];              // j-rows

    const int i0  = blockIdx.y * TILE;
    const int j0  = blockIdx.x * TILE;
    const int tid = threadIdx.x;

    // ---- cooperative staging: 128x128 f16 tile = 2048 x 16B chunks, 8/thread ----
    #pragma unroll
    for (int c = 0; c < 8; ++c) {
        const int chunk = c * 256 + tid;
        const int row   = chunk >> 4;
        const int col   = (chunk & 15) * 8;  // half offset
        *(v8h*)&Bs[row][col] = *(const v8h*)(nrm16 + (size_t)(j0 + row) * H_N + col);
    }
    ((v4f*)&logp_s[0][0])[tid] = ((const v4f*)(logp + (size_t)i0 * E_N))[tid];
    ((v4f*)&p_s[0][0])[tid]    = ((const v4f*)(p    + (size_t)j0 * E_N))[tid];
    if (tid < TILE) a_s[tid] = a[j0 + tid];
    __syncthreads();

    const int wave = tid >> 5;        // 0..7 -> 16-row strip of the tile
    const int lane = tid & 31;
    const int m0w  = wave * 16;
    const int lrow = lane & 15;
    const int hi   = lane >> 4;

    // ---- sim A fragments (16x32 f16 per k-step) straight from global, held in regs ----
    // layout (ISA 7.12.2): lane l holds row M=l%16; K = (l/16)*8 + [0..7] and +16..+23
    v16h afrag[4];
    {
        const _Float16* gA = nrm16 + (size_t)(i0 + m0w + lrow) * H_N;
        const int kb = hi * 8;
        #pragma unroll
        for (int ks = 0; ks < 4; ++ks) {
            const v8h lo = *(const v8h*)(gA + ks * 32 + kb);
            const v8h hh = *(const v8h*)(gA + ks * 32 + kb + 16);
            v16h af;
            #pragma unroll
            for (int e = 0; e < 8; ++e) { af[e] = lo[e]; af[8 + e] = hh[e]; }
            afrag[ks] = af;
        }
    }

    // ---- KL A fragments (f32 16x16x4, K=8 in 2 steps), wave-invariant: hoisted ----
    // 32-bit A 16x4 layout: lane l holds row M=l%16; V0 = K=(l/16)*2, V1 = K=(l/16)*2+1
    v2f klA[2];
    {
        const int arow = m0w + lrow;
        #pragma unroll
        for (int ks = 0; ks < 2; ++ks)
            klA[ks] = *(const v2f*)&logp_s[arow][ks * 4 + hi * 2];
    }

    float local_kl = 0.f;
    unsigned int local_cnt = 0;       // wave-uniform (ballot/popc), no lane reduction needed

    // two independent column-tiles per iteration: fills WMMA D->C chain latency and
    // gives the scheduler work to cover the WMMA->VALU hazard window
    #pragma unroll 1
    for (int n0 = 0; n0 < TILE; n0 += 32) {
        const int jcA = n0 + lrow;
        const int jcB = n0 + 16 + lrow;

        v8f acc0, acc1, dot0, dot1;
        #pragma unroll
        for (int e = 0; e < 8; ++e) { acc0[e] = 0.f; acc1[e] = 0.f; dot0[e] = 0.f; dot1[e] = 0.f; }

        #pragma unroll
        for (int ks = 0; ks < 4; ++ks) {
            // B fragment (32x16): lane l holds column N=l%16; K = ks*32 + (l/16)*8 + [0..7] and +16..+23
            const int kb2 = ks * 32 + hi * 8;
            const v8h a0 = *(const v8h*)&Bs[jcA][kb2];
            const v8h a1 = *(const v8h*)&Bs[jcA][kb2 + 16];
            const v8h b0 = *(const v8h*)&Bs[jcB][kb2];
            const v8h b1 = *(const v8h*)&Bs[jcB][kb2 + 16];
            v16h bfA, bfB;
            #pragma unroll
            for (int e = 0; e < 8; ++e) {
                bfA[e] = a0[e]; bfA[8 + e] = a1[e];
                bfB[e] = b0[e]; bfB[8 + e] = b1[e];
            }
            acc0 = __builtin_amdgcn_wmma_f32_16x16x32_f16(
                false, afrag[ks], false, bfA, (short)0, acc0, false, false);
            acc1 = __builtin_amdgcn_wmma_f32_16x16x32_f16(
                false, afrag[ks], false, bfB, (short)0, acc1, false, false);
        }

        // kl_dot tiles: logp(16x8) @ p^T(8x16), f32 WMMA, K=4 per step
        // 32-bit B 4x16 layout (symmetric to A): lane l holds N=l%16; V0/V1 = K=(l/16)*2, +1
        #pragma unroll
        for (int ks = 0; ks < 2; ++ks) {
            const v2f kbA = *(const v2f*)&p_s[jcA][ks * 4 + hi * 2];
            const v2f kbB = *(const v2f*)&p_s[jcB][ks * 4 + hi * 2];
            dot0 = __builtin_amdgcn_wmma_f32_16x16x4_f32(
                false, klA[ks], false, kbA, (short)0, dot0, false, false);
            dot1 = __builtin_amdgcn_wmma_f32_16x16x4_f32(
                false, klA[ks], false, kbB, (short)0, dot1, false, false);
        }

        // ---- branchless masked accumulation: kl(i,j) = a[j] - logp[i].p[j] ----
        // Diagonal (i==j) is NOT excluded here: sim(i,i)~1 always passes, and
        // kl(i,i) = a[i] - logp[i].p[i] ~ 0 by construction, so it adds exactly
        // B_N to the count (corrected in finalize) and only rounding noise to kl.
        const float ajA = a_s[jcA];
        const float ajB = a_s[jcB];
        #pragma unroll
        for (int rr = 0; rr < 8; ++rr) {
            const bool mA = acc0[rr] > SIM_T;
            const bool mB = acc1[rr] > SIM_T;
            local_kl  += mA ? (ajA - dot0[rr]) : 0.f;     // v_cmp + v_cndmask + v_add
            local_kl  += mB ? (ajB - dot1[rr]) : 0.f;
            local_cnt += (unsigned int)__popcll(__ballot(mA));  // scalar s_bcnt path
            local_cnt += (unsigned int)__popcll(__ballot(mB));
        }
    }

    #pragma unroll
    for (int m = 16; m > 0; m >>= 1) local_kl += __shfl_xor(local_kl, m, 32);
    if (lane == 0) {
        atomicAdd(kl_acc, (double)local_kl);
        atomicAdd(cnt_acc, (unsigned long long)local_cnt);
    }
}

__global__ void finalize_kernel(const float* __restrict__ acc,
                                const double* __restrict__ kl_acc,
                                const unsigned long long* __restrict__ cnt_acc,
                                const float* __restrict__ temperature,
                                float* __restrict__ out)
{
    if (threadIdx.x == 0) {
        const float invB = 1.0f / (float)B_N;
        const float task = acc[0] * invB;
        const float eff_loss = W_EFF * (acc[1] * invB);
        const float ent_loss = W_ENT * (-(acc[2] * invB));

        float usage[E_N], mean = 0.f;
        #pragma unroll
        for (int e = 0; e < E_N; ++e) { usage[e] = acc[3 + e] * invB; mean += usage[e]; }
        mean *= (1.0f / E_N);
        float var = 0.f;
        #pragma unroll
        for (int e = 0; e < E_N; ++e) { const float d = usage[e] - mean; var += d * d; }
        var *= (1.0f / (E_N - 1));               // unbiased (ddof=1)
        const float lb_loss = W_LB * var * (float)(E_N * E_N);

        // subtract the B_N diagonal pairs that were counted but must be excluded
        const long long n = (long long)(*cnt_acc) - (long long)B_N;
        const float cons_loss = (n > 0)
            ? (float)((double)W_CONS * (*kl_acc / (double)n)) : 0.f;

        const float dt = temperature[0] - 1.0f;
        const float temp_loss = W_TEMP * dt * dt;

        out[0] = task + lb_loss + eff_loss + cons_loss + ent_loss + temp_loss;
    }
}

extern "C" void kernel_launch(void* const* d_in, const int* in_sizes, int n_in,
                              void* d_out, int out_size, void* d_ws, size_t ws_size,
                              hipStream_t stream) {
    const float* logits  = (const float*)d_in[0];
    const int*   targets = (const int*)  d_in[1];
    const float* rp      = (const float*)d_in[2];
    const float* emb     = (const float*)d_in[3];
    const float* temp    = (const float*)d_in[4];

    unsigned char* ws = (unsigned char*)d_ws;
    float*  acc   = (float*)ws;
    double* klsum = (double*)(ws + 48);
    unsigned long long* cnt = (unsigned long long*)(ws + 56);
    float*     a_arr    = (float*)(ws + 256);
    float*     logp_arr = (float*)(ws + 33024);
    float*     p_arr    = (float*)(ws + 295168);
    _Float16*  nrm16    = (_Float16*)(ws + 557312);

    hipLaunchKernelGGL(zero_acc_kernel, dim3(1), dim3(32), 0, stream, (unsigned int*)ws);
    hipLaunchKernelGGL(prep_kernel, dim3((B_N * 32) / 256), dim3(256), 0, stream,
                       logits, targets, rp, emb, acc, a_arr, logp_arr, p_arr, nrm16);
    hipLaunchKernelGGL(cons_kernel, dim3(B_N / TILE, B_N / TILE), dim3(256), 0, stream,
                       nrm16, logp_arr, p_arr, a_arr, klsum, cnt);
    hipLaunchKernelGGL(finalize_kernel, dim3(1), dim3(1), 0, stream,
                       acc, klsum, cnt, temp, (float*)d_out);
}
